// C3WithGLCM_4243427688613
// MI455X (gfx1250) — compile-verified, matrix operand order checked by reading, not run on hardware
//
#include <hip/hip_runtime.h>
#include <hip/hip_bf16.h>

// ---------------------------------------------------------------------------
// CDNA5 (gfx1250): GLCM features + C3 conv block via bf16 WMMA.
// All GEMM tiles staged in LDS row-major per-fragment (16B-aligned rows) so
// both A and B fragments load as 2x ds_load_b128. Xc is kept pixel-major so
// the cv1/cv2 activation tiles can be staged with async global->LDS copies.
// ---------------------------------------------------------------------------

typedef __attribute__((ext_vector_type(16))) __bf16 v16bf;
typedef __attribute__((ext_vector_type(8)))  float  v8f;

#define NP 25600      // B*H*W pixels
#define HW 6400       // H*W
#define WD 80
#define KXC 96        // padded K for concat(x,g): 68 -> 96

#if __has_builtin(__builtin_amdgcn_global_load_async_to_lds_b128) && \
    __has_builtin(__builtin_amdgcn_s_wait_asynccnt)
#define HAVE_ASYNC_LDS 1
// Parameter types per hipcc diagnostic: v4i in AS1 (global) and AS3 (LDS).
typedef __attribute__((__vector_size__(4 * sizeof(int)))) int vi4;
typedef vi4 __attribute__((address_space(1)))* gvi4p;
typedef vi4 __attribute__((address_space(3)))* lvi4p;
#else
#define HAVE_ASYNC_LDS 0
#endif

__device__ __forceinline__ float silu_f(float v) {
    return v * (1.0f / (1.0f + __expf(-v)));
}

// ---- WMMA fragment loader --------------------------------------------------
// Row-major 16x32 bf16 tile in LDS (row stride ldk halves, rows 16B aligned).
// Lane l holds row (rowBase + l%16); k-halves: lanes 0-15 -> K {0..7},{16..23},
// lanes 16-31 -> K {8..15},{24..31}; frag elem 2r/2r+1 = VGPR r lo/hi.
// Used for A (row = M) and, with the mirrored B layout, for B^T (row = N).
__device__ __forceinline__ v16bf load_frag16(const __bf16* lds, int ldk,
                                             int rowBase, int lane) {
    const int r    = rowBase + (lane & 15);
    const int koff = (lane >> 4) * 8;
    const __bf16* row = lds + r * ldk;
    v16bf f;
#pragma unroll
    for (int e = 0; e < 8; ++e)  f[e] = row[koff + e];        // 16B contiguous
#pragma unroll
    for (int e = 8; e < 16; ++e) f[e] = row[8 + koff + e];    // K = 16+koff+..
    return f;
}

__device__ __forceinline__ v8f wmma_bf16(v16bf a, v16bf b, v8f c) {
    return __builtin_amdgcn_wmma_f32_16x16x32_bf16(
        /*neg_a=*/false, a, /*neg_b=*/false, b,
        /*c_mod=*/(short)0, c, /*reuse_a=*/false, /*reuse_b=*/false);
}

// ---------------------------------------------------------------------------
// Kernel 0: zero GLCM accumulator
// ---------------------------------------------------------------------------
__global__ __launch_bounds__(256) void zero_g_kernel(float* __restrict__ g, int n) {
    int i = blockIdx.x * 256 + threadIdx.x;
    if (i < n) g[i] = 0.0f;
}

// ---------------------------------------------------------------------------
// Kernel 1: GLCM features, one thread per (b,c,h,w); LDS byte histograms,
//           channel-average folded via global f32 atomics.
// ---------------------------------------------------------------------------
__global__ __launch_bounds__(256) void glcm_kernel(const float* __restrict__ x,
                                                   float* __restrict__ g) {
    __shared__ unsigned char hist[256 * 68];
    const int tid = threadIdx.x;
    unsigned char* h = &hist[tid * 68];
    unsigned int* h32 = (unsigned int*)h;          // 68 bytes = 17 dwords
#pragma unroll
    for (int i = 0; i < 17; ++i) h32[i] = 0u;

    const int gid = blockIdx.x * 256 + tid;        // exactly B*C*HW threads
    const int bc  = gid / HW;
    const int hw  = gid - bc * HW;
    const int h0  = hw / WD, w0 = hw - (hw / WD) * WD;
    const float* base = x + bc * HW;

    // quantize 3x3 edge-clamped patch to 8 levels
    int q[9];
#pragma unroll
    for (int dr = 0; dr < 3; ++dr) {
#pragma unroll
        for (int dc = 0; dc < 3; ++dc) {
            int r = h0 + dr - 1; r = r < 0 ? 0 : (r > 79 ? 79 : r);
            int c = w0 + dc - 1; c = c < 0 ? 0 : (c > 79 ? 79 : c);
            int qq = (int)(base[r * WD + c] * 7.0f);
            q[dr * 3 + dc] = qq < 0 ? 0 : (qq > 7 ? 7 : qq);
        }
    }

    // 20 co-occurrence pairs over 4 directions
    const signed char I1[20] = {0,1,3,4,6,7, 0,1,2,3,4,5, 0,1,3,4, 1,2,4,5};
    const signed char I2[20] = {1,2,4,5,7,8, 3,4,5,6,7,8, 4,5,7,8, 3,4,6,7};
#pragma unroll
    for (int k = 0; k < 20; ++k)
        h[q[(int)I1[k]] * 8 + q[(int)I2[k]]] += 1;

    // features; p[bin] = count/20 exactly (hist sum is always 5 after /4)
    float contrast = 0.f, energy = 0.f, entropy = 0.f, homog = 0.f;
#pragma unroll
    for (int bin = 0; bin < 64; ++bin) {
        int c = h[bin];
        if (c) {
            float p = (float)c * 0.05f;
            int i = bin >> 3, j = bin & 7;
            float d = (float)(i - j);
            contrast += p * d * d;
            energy   += p * p;
            entropy  -= p * __logf(p + 1e-6f);
            homog    += p * (1.0f / (1.0f + fabsf(d)));
        }
    }

    const int b = bc >> 6;                 // / C
    float* gp = g + b * 4 * HW + hw;
    const float invC = 1.0f / 64.0f;
    atomicAdd(gp + 0 * HW, contrast * invC);
    atomicAdd(gp + 1 * HW, energy   * invC);
    atomicAdd(gp + 2 * HW, entropy  * invC);
    atomicAdd(gp + 3 * HW, homog    * invC);
}

// ---------------------------------------------------------------------------
// Kernel 2: build XcT = concat(x,g) as bf16, pixel-major [p][96], K padded
//           68->96 with zeros. Row = 192 B, 16B aligned -> async-LDS friendly.
// ---------------------------------------------------------------------------
__global__ __launch_bounds__(256) void build_xc_kernel(const float* __restrict__ x,
                                                       const float* __restrict__ g,
                                                       __bf16* __restrict__ XcT) {
    const int idx = blockIdx.x * 256 + threadIdx.x;   // KXC * NP threads
    const int p = idx / KXC;
    const int k = idx - p * KXC;
    const int b = p / HW, hw = p - (p / HW) * HW;
    float v = 0.0f;
    if (k < 64)       v = x[b * 64 * HW + k * HW + hw];
    else if (k < 68)  v = g[b * 4 * HW + (k - 64) * HW + hw];
    XcT[idx] = (__bf16)v;
}

// ---------------------------------------------------------------------------
// Kernel 3: cv1 + cv2 fused 1x1 GEMM (M=64 stacked, K=68 padded 96, Nmacro=32)
//           B tile staged transposed [N][K]; async global->LDS when available.
// ---------------------------------------------------------------------------
__global__ __launch_bounds__(256) void gemm_cv12_kernel(
    const float* __restrict__ w1, const float* __restrict__ s1, const float* __restrict__ b1,
    const float* __restrict__ w2, const float* __restrict__ s2, const float* __restrict__ b2,
    const __bf16* __restrict__ XcT, float* __restrict__ A0, float* __restrict__ Bb) {
    __shared__ __bf16 ldsA[64][40];   // [M][K], rows 80B (16B aligned)
    __shared__ __bf16 ldsB[32][40];   // [N][K]
    const int tid  = threadIdx.x;
    const int lane = tid & 31, wave = tid >> 5;
    const int n0 = blockIdx.x * 32;
    const int mBase = (wave & 3) * 16, nBase = (wave >> 2) * 16;

    v8f acc = {};
#pragma unroll
    for (int kc = 0; kc < 3; ++kc) {
        const int k0 = kc * 32;
        for (int i = tid; i < 64 * 32; i += 256) {          // weights (zero-pad K>=68)
            int m = i >> 5, kk = i & 31, k = k0 + kk;
            float v = 0.0f;
            if (k < 68) v = (m < 32) ? w1[m * 68 + k] : w2[(m - 32) * 68 + k];
            ldsA[m][kk] = (__bf16)v;
        }
        if (tid < 128) {                                    // 32 rows x 64B
            int nn = tid >> 2, part = tid & 3;
            const __bf16* gsrc = XcT + (n0 + nn) * KXC + k0 + part * 8;
            __bf16* ldst = &ldsB[nn][part * 8];
#if HAVE_ASYNC_LDS
            __builtin_amdgcn_global_load_async_to_lds_b128(
                (gvi4p)(void*)gsrc, (lvi4p)(void*)ldst, 0, 0);
#else
            *(uint4*)ldst = *(const uint4*)gsrc;
#endif
        }
        if (kc + 1 < 3) __builtin_prefetch(XcT + n0 * KXC + k0 + 32, 0, 1);
#if HAVE_ASYNC_LDS
        __builtin_amdgcn_s_wait_asynccnt(0);
#endif
        __syncthreads();
        v16bf a = load_frag16(&ldsA[0][0], 40, mBase, lane);
        v16bf b = load_frag16(&ldsB[0][0], 40, nBase, lane);
        acc = wmma_bf16(a, b, acc);
        __syncthreads();
    }

    const int p = n0 + nBase + (lane & 15);
#pragma unroll
    for (int e = 0; e < 8; ++e) {
        int m = mBase + e + ((lane >> 4) << 3);
        float v = acc[e];
        if (m < 32) { v = silu_f(v * s1[m] + b1[m]);          A0[m * NP + p] = v; }
        else        { int m2 = m - 32;
                      v = silu_f(v * s2[m2] + b2[m2]);        Bb[m2 * NP + p] = v; }
    }
}

// ---------------------------------------------------------------------------
// Kernel 4: m1 1x1 GEMM 32->16 (K=32, single chunk, N macro=128)
// ---------------------------------------------------------------------------
__global__ __launch_bounds__(256) void gemm_m1_kernel(
    const float* __restrict__ w, const float* __restrict__ s, const float* __restrict__ b,
    const float* __restrict__ A0, float* __restrict__ Y1) {
    __shared__ __bf16 ldsA[16][40];    // [M][K]
    __shared__ __bf16 ldsB[128][40];   // [N][K]
    const int tid  = threadIdx.x;
    const int lane = tid & 31, wave = tid >> 5;
    const int n0 = blockIdx.x * 128;
    const int nBase = wave * 16;

    for (int i = tid; i < 16 * 32; i += 256) {
        int m = i >> 5, kk = i & 31;
        ldsA[m][kk] = (__bf16)w[m * 32 + kk];
    }
    for (int i = tid; i < 32 * 128; i += 256) {     // nn fast: coalesced reads
        int kk = i >> 7, nn = i & 127;
        ldsB[nn][kk] = (__bf16)A0[kk * NP + n0 + nn];
    }
    __syncthreads();

    v8f acc = {};
    v16bf af = load_frag16(&ldsA[0][0], 40, 0, lane);
    v16bf bf = load_frag16(&ldsB[0][0], 40, nBase, lane);
    acc = wmma_bf16(af, bf, acc);

    const int p = n0 + nBase + (lane & 15);
#pragma unroll
    for (int e = 0; e < 8; ++e) {
        int m = e + ((lane >> 4) << 3);
        Y1[m * NP + p] = silu_f(acc[e] * s[m] + b[m]);
    }
}

// ---------------------------------------------------------------------------
// Kernel 5: m2 3x3 GEMM 16->32 (im2col K=144 padded 160) + residual with A0
// ---------------------------------------------------------------------------
__global__ __launch_bounds__(256) void gemm_m2_kernel(
    const float* __restrict__ w, const float* __restrict__ s, const float* __restrict__ b,
    const float* __restrict__ Y1, const float* __restrict__ A0, float* __restrict__ A2) {
    __shared__ __bf16 ldsA[32][40];   // [M][K]
    __shared__ __bf16 ldsB[64][40];   // [N][K]
    const int tid  = threadIdx.x;
    const int lane = tid & 31, wave = tid >> 5;
    const int n0 = blockIdx.x * 64;
    const int mBase = (wave & 1) * 16, nBase = (wave >> 1) * 16;

    v8f acc = {};
#pragma unroll 1
    for (int kc = 0; kc < 5; ++kc) {
        const int k0 = kc * 32;
        for (int i = tid; i < 32 * 32; i += 256) {          // weights: k = tap*16+ic
            int m = i >> 5, kk = i & 31, k = k0 + kk;
            float v = 0.0f;
            if (k < 144) {
                int ic = k & 15, tap = k >> 4;
                v = w[m * 144 + ic * 9 + tap];
            }
            ldsA[m][kk] = (__bf16)v;
        }
        for (int i = tid; i < 32 * 64; i += 256) {          // im2col, zero pad
            int kk = i >> 6, nn = i & 63, k = k0 + kk;
            float v = 0.0f;
            if (k < 144) {
                int ic = k & 15, tap = k >> 4;
                int dh = tap / 3 - 1, dw = tap - (tap / 3) * 3 - 1;
                int p = n0 + nn;
                int hw = p - (p / HW) * HW;
                int hh = hw / WD + dh, ww = hw - (hw / WD) * WD + dw;
                if (hh >= 0 && hh < WD && ww >= 0 && ww < WD)
                    v = Y1[ic * NP + (p - hw) + hh * WD + ww];
            }
            ldsB[nn][kk] = (__bf16)v;
        }
        __syncthreads();
        v16bf af = load_frag16(&ldsA[0][0], 40, mBase, lane);
        v16bf bf = load_frag16(&ldsB[0][0], 40, nBase, lane);
        acc = wmma_bf16(af, bf, acc);
        __syncthreads();
    }

    const int p = n0 + nBase + (lane & 15);
#pragma unroll
    for (int e = 0; e < 8; ++e) {
        int m = mBase + e + ((lane >> 4) << 3);
        float v = silu_f(acc[e] * s[m] + b[m]);
        A2[m * NP + p] = A0[m * NP + p] + v;               // bottleneck shortcut
    }
}

// ---------------------------------------------------------------------------
// Kernel 6: cv3 1x1 GEMM 64->64 over concat(A2, Bb); writes d_out (B,64,H,W)
// ---------------------------------------------------------------------------
__global__ __launch_bounds__(256) void gemm_cv3_kernel(
    const float* __restrict__ w, const float* __restrict__ s, const float* __restrict__ b,
    const float* __restrict__ A2, const float* __restrict__ Bb, float* __restrict__ out) {
    __shared__ __bf16 ldsA[64][40];   // [M][K]
    __shared__ __bf16 ldsB[32][40];   // [N][K]
    const int tid  = threadIdx.x;
    const int lane = tid & 31, wave = tid >> 5;
    const int n0 = blockIdx.x * 32;
    const int mBase = (wave & 3) * 16, nBase = (wave >> 2) * 16;

    v8f acc = {};
#pragma unroll
    for (int kc = 0; kc < 2; ++kc) {
        const int k0 = kc * 32;
        for (int i = tid; i < 64 * 32; i += 256) {
            int m = i >> 5, kk = i & 31;
            ldsA[m][kk] = (__bf16)w[m * 64 + k0 + kk];
        }
        for (int i = tid; i < 32 * 32; i += 256) {
            int kk = i >> 5, nn = i & 31, k = k0 + kk;
            float v = (k < 32) ? A2[k * NP + n0 + nn] : Bb[(k - 32) * NP + n0 + nn];
            ldsB[nn][kk] = (__bf16)v;
        }
        __syncthreads();
        v16bf af = load_frag16(&ldsA[0][0], 40, mBase, lane);
        v16bf bf = load_frag16(&ldsB[0][0], 40, nBase, lane);
        acc = wmma_bf16(af, bf, acc);
        __syncthreads();
    }

    const int p = n0 + nBase + (lane & 15);
    const int bb = p / HW, hw = p - bb * HW;
#pragma unroll
    for (int e = 0; e < 8; ++e) {
        int m = mBase + e + ((lane >> 4) << 3);
        out[bb * 64 * HW + m * HW + hw] = silu_f(acc[e] * s[m] + b[m]);
    }
}

// ---------------------------------------------------------------------------
// Host-side launcher
// ---------------------------------------------------------------------------
extern "C" void kernel_launch(void* const* d_in, const int* in_sizes, int n_in,
                              void* d_out, int out_size, void* d_ws, size_t ws_size,
                              hipStream_t stream) {
    const float* x     = (const float*)d_in[0];
    const float* cv1_w = (const float*)d_in[1];
    const float* cv1_s = (const float*)d_in[2];
    const float* cv1_b = (const float*)d_in[3];
    const float* cv2_w = (const float*)d_in[4];
    const float* cv2_s = (const float*)d_in[5];
    const float* cv2_b = (const float*)d_in[6];
    const float* m1_w  = (const float*)d_in[7];
    const float* m1_s  = (const float*)d_in[8];
    const float* m1_b  = (const float*)d_in[9];
    const float* m2_w  = (const float*)d_in[10];
    const float* m2_s  = (const float*)d_in[11];
    const float* m2_b  = (const float*)d_in[12];
    const float* cv3_w = (const float*)d_in[13];
    const float* cv3_s = (const float*)d_in[14];
    const float* cv3_b = (const float*)d_in[15];

    char* wsb = (char*)d_ws;
    float*  g   = (float*)(wsb + 0);            // 102400 f32
    __bf16* XcT = (__bf16*)(wsb + 409600);      // NP*96 bf16, pixel-major
    float*  A0  = (float*)(wsb + 5324800);      // 32*25600 f32
    float*  Bb  = (float*)(wsb + 8601600);      // 32*25600 f32
    float*  Y1  = (float*)(wsb + 11878400);     // 16*25600 f32
    float*  A2  = (float*)(wsb + 13516800);     // 32*25600 f32
    float*  out = (float*)d_out;

    zero_g_kernel<<<400, 256, 0, stream>>>(g, 4 * 4 * HW);
    glcm_kernel<<<6400, 256, 0, stream>>>(x, g);
    build_xc_kernel<<<9600, 256, 0, stream>>>(x, g, XcT);
    gemm_cv12_kernel<<<800, 256, 0, stream>>>(cv1_w, cv1_s, cv1_b,
                                              cv2_w, cv2_s, cv2_b, XcT, A0, Bb);
    gemm_m1_kernel<<<200, 256, 0, stream>>>(m1_w, m1_s, m1_b, A0, Y1);
    gemm_m2_kernel<<<400, 256, 0, stream>>>(m2_w, m2_s, m2_b, Y1, A0, A2);
    gemm_cv3_kernel<<<800, 256, 0, stream>>>(cv3_w, cv3_s, cv3_b, A2, Bb, out);
}
